// TransformerBlockPHGate_37546604101859
// MI455X (gfx1250) — matrix operand order repaired and checked by reading, not compiled, not run on hardware
//
#include <hip/hip_runtime.h>
#include <hip/hip_bf16.h>
#include <math.h>

#define E_  1024
#define H_  16
#define D_  64
#define FF_ 4096
#define B_  2
#define S_  2048
#define M_  (B_ * S_)   // 4096 token rows

typedef __attribute__((ext_vector_type(16))) __bf16 v16bf;
typedef __attribute__((ext_vector_type(8)))  __bf16 v8bf;
typedef __attribute__((ext_vector_type(8)))  float  v8f;

#define WMMA_BF16(a, b, c) \
  __builtin_amdgcn_wmma_f32_16x16x32_bf16(false, (a), false, (b), (short)0, (c), false, false)

// ---------------------------------------------------------------------------
// Fragment loaders per CDNA5 ISA 7.12.2 VGPR layouts (wave32)
// A (16x32 bf16, row-major, stride lda): lane m=L%16, chunks at k=(L>>4)*8, +16
// B (32x16 bf16) from Bt (N x K row-major): lane n=L%16, chunks at k=(L>>4)*8, +16
// C/D (16x16 f32): VGPR r holds row m = r + (L>>4)*8, col n = L%16
// ---------------------------------------------------------------------------
__device__ inline v16bf load_afrag(const __bf16* A, int lda, int lane, int kk) {
  int m = lane & 15, k0 = (lane >> 4) * 8;
  const __bf16* p = A + (size_t)m * lda + kk + k0;
  union { v16bf v; v8bf h[2]; } u;
  u.h[0] = *(const v8bf*)(p);
  u.h[1] = *(const v8bf*)(p + 16);
  return u.v;
}

__device__ inline v16bf load_bfrag(const __bf16* Bt, int ldb, int lane, int kk) {
  int n = lane & 15, k0 = (lane >> 4) * 8;
  const __bf16* p = Bt + (size_t)n * ldb + kk + k0;
  union { v16bf v; v8bf h[2]; } u;
  u.h[0] = *(const v8bf*)(p);
  u.h[1] = *(const v8bf*)(p + 16);
  return u.v;
}

// ---------------------------------------------------------------------------
// Elementwise f32 -> bf16
// ---------------------------------------------------------------------------
__global__ void cvt_bf16_kernel(const float* __restrict__ in, __bf16* __restrict__ out, int n) {
  int i = blockIdx.x * blockDim.x + threadIdx.x;
  if (i < n) out[i] = (__bf16)in[i];
}

// in: R x C f32 (row-major) -> out: C x R bf16 (row-major), i.e. out = in^T
__global__ void __launch_bounds__(256) transpose_cvt_kernel(const float* __restrict__ in,
                                                            __bf16* __restrict__ out,
                                                            int R, int C) {
  __shared__ float tile[32][33];
  int c0 = blockIdx.x * 32, r0 = blockIdx.y * 32;
  int tx = threadIdx.x & 31, ty = threadIdx.x >> 5;  // 8 rows of 32
  for (int i = ty; i < 32; i += 8) {
    int r = r0 + i, c = c0 + tx;
    tile[i][tx] = (r < R && c < C) ? in[(size_t)r * C + c] : 0.f;
  }
  __syncthreads();
  for (int i = ty; i < 32; i += 8) {
    int c = c0 + i, r = r0 + tx;
    if (c < C && r < R) out[(size_t)c * R + r] = (__bf16)tile[tx][i];
  }
}

// value (B,S,E) f32 -> vt (B,H,D,S) bf16 (per-head transposed V for P@V WMMA)
__global__ void v_transpose_kernel(const float* __restrict__ v, __bf16* __restrict__ vt) {
  size_t i = (size_t)blockIdx.x * blockDim.x + threadIdx.x;
  if (i >= (size_t)B_ * E_ * S_) return;
  int s = (int)(i % S_); size_t j = i / S_;
  int d = (int)(j % D_); j /= D_;
  int h = (int)(j % H_); int b = (int)(j / H_);
  vt[i] = (__bf16)v[((size_t)b * S_ + s) * E_ + h * D_ + d];
}

// ---------------------------------------------------------------------------
// x_avg = mean over seq of query : (B,E)
// ---------------------------------------------------------------------------
__global__ void colmean_kernel(const float* __restrict__ q, float* __restrict__ xavg) {
  int i = blockIdx.x * blockDim.x + threadIdx.x;
  if (i >= B_ * E_) return;
  int e = i % E_, b = i / E_;
  const float* p = q + (size_t)b * S_ * E_ + e;
  float s = 0.f;
  for (int t = 0; t < S_; ++t) s += p[(size_t)t * E_];
  xavg[i] = s * (1.0f / S_);
}

// gate = sigmoid(x_avg @ Wg + bg) : (B,E)
__global__ void gate_kernel(const float* __restrict__ xavg, const float* __restrict__ Wg,
                            const float* __restrict__ bg, float* __restrict__ gate) {
  int i = blockIdx.x * blockDim.x + threadIdx.x;
  if (i >= B_ * E_) return;
  int eo = i % E_, b = i / E_;
  const float* x = xavg + (size_t)b * E_;
  float s = bg[eo];
  for (int e = 0; e < E_; ++e) s += x[e] * Wg[(size_t)e * E_ + eo];
  gate[i] = 1.0f / (1.0f + __expf(-s));
}

// ---------------------------------------------------------------------------
// Flash-attention: one wave per 16-row query tile. Streams 32 keys per step.
// energy computed with wmma bf16, masked (causal, BEFORE 1/8 scaling as in the
// reference), online softmax, P packed via LDS into an A-fragment, P@Vt wmma.
// ---------------------------------------------------------------------------
__global__ void __launch_bounds__(128) attn_kernel(const __bf16* __restrict__ qb,
                                                   const __bf16* __restrict__ kb,
                                                   const __bf16* __restrict__ vt,
                                                   __bf16* __restrict__ ao) {
  __shared__ __bf16 ldsP[4][16 * 32];   // per-wave 16x32 bf16 staging
  const int wave = threadIdx.x >> 5;
  const int lane = threadIdx.x & 31;
  const int gw   = blockIdx.x * 4 + wave;
  const int NQT  = S_ / 16;
  const int qt = gw % NQT;
  const int bh = gw / NQT;
  const int h = bh % H_, b = bh / H_;
  const int q0 = qt * 16;
  const int hlf = lane >> 4, nI = lane & 15;

  const __bf16* Q  = qb + (size_t)b * S_ * E_ + h * D_;
  const __bf16* Kp = kb + (size_t)b * S_ * E_ + h * D_;
  const __bf16* Vp = vt + (size_t)bh * D_ * S_;
  __bf16* lp = ldsP[wave];

  // Q fragments for this tile (K-dim = head_dim = 64 -> two 32-wide chunks)
  v16bf aq0 = load_afrag(Q + (size_t)q0 * E_, E_, lane, 0);
  v16bf aq1 = load_afrag(Q + (size_t)q0 * E_, E_, lane, 32);

  v8f o0 = {}, o1 = {}, o2 = {}, o3 = {};
  float mrow[8], srow[8];
#pragma unroll
  for (int r = 0; r < 8; ++r) { mrow[r] = -3.0e38f; srow[r] = 0.f; }

  for (int k0 = 0; k0 < q0 + 16; k0 += 32) {
    // ----- energy block 16x32 (two 16x16 tiles) -----
    v8f e0 = {}, e1 = {};
    {
      v16bf bk0 = load_bfrag(Kp + (size_t)k0 * E_, E_, lane, 0);
      v16bf bk1 = load_bfrag(Kp + (size_t)k0 * E_, E_, lane, 32);
      e0 = WMMA_BF16(aq0, bk0, e0);
      e0 = WMMA_BF16(aq1, bk1, e0);
      v16bf bk2 = load_bfrag(Kp + (size_t)(k0 + 16) * E_, E_, lane, 0);
      v16bf bk3 = load_bfrag(Kp + (size_t)(k0 + 16) * E_, E_, lane, 32);
      e1 = WMMA_BF16(aq0, bk2, e1);
      e1 = WMMA_BF16(aq1, bk3, e1);
    }

    // ----- mask (before scale), online softmax update -----
    float p0[8], p1[8];
#pragma unroll
    for (int r = 0; r < 8; ++r) {
      int m = q0 + r + hlf * 8;
      float x0 = (k0 + nI      <= m) ? e0[r] * 0.125f : -3.0e38f;
      float x1 = (k0 + 16 + nI <= m) ? e1[r] * 0.125f : -3.0e38f;
      float bm = fmaxf(x0, x1);
#pragma unroll
      for (int sft = 8; sft >= 1; sft >>= 1) bm = fmaxf(bm, __shfl_xor(bm, sft, 32));
      float nm = fmaxf(mrow[r], bm);
      float al = __expf(mrow[r] - nm);
      p0[r] = __expf(x0 - nm);
      p1[r] = __expf(x1 - nm);
      float ps = p0[r] + p1[r];
#pragma unroll
      for (int sft = 8; sft >= 1; sft >>= 1) ps += __shfl_xor(ps, sft, 32);
      srow[r] = srow[r] * al + ps;
      mrow[r] = nm;
      o0[r] *= al; o1[r] *= al; o2[r] *= al; o3[r] *= al;
    }

    // ----- pack P (C-layout regs) -> LDS -> reload as A fragment -----
#pragma unroll
    for (int r = 0; r < 8; ++r) {
      lp[(r + hlf * 8) * 32 + nI]      = (__bf16)p0[r];
      lp[(r + hlf * 8) * 32 + 16 + nI] = (__bf16)p1[r];
    }
    asm volatile("s_wait_dscnt 0" ::: "memory");
    v16bf pa;
    {
      union { v16bf v; v8bf hh[2]; } u;
      const __bf16* pr = lp + (size_t)(lane & 15) * 32 + hlf * 8;
      u.hh[0] = *(const v8bf*)(pr);
      u.hh[1] = *(const v8bf*)(pr + 16);
      pa = u.v;
    }

    // ----- O += P @ V^T (4 d-tiles of 16) -----
    o0 = WMMA_BF16(pa, load_bfrag(Vp + (size_t) 0 * S_, S_, lane, k0), o0);
    o1 = WMMA_BF16(pa, load_bfrag(Vp + (size_t)16 * S_, S_, lane, k0), o1);
    o2 = WMMA_BF16(pa, load_bfrag(Vp + (size_t)32 * S_, S_, lane, k0), o2);
    o3 = WMMA_BF16(pa, load_bfrag(Vp + (size_t)48 * S_, S_, lane, k0), o3);
  }

  // ----- normalize, write concat-head output (bf16, feeds Wo GEMM) -----
#pragma unroll
  for (int r = 0; r < 8; ++r) {
    float inv = 1.0f / srow[r];
    int srw = q0 + r + hlf * 8;
    __bf16* dst = ao + ((size_t)b * S_ + srw) * E_ + h * D_ + nI;
    dst[0]  = (__bf16)(o0[r] * inv);
    dst[16] = (__bf16)(o1[r] * inv);
    dst[32] = (__bf16)(o2[r] * inv);
    dst[48] = (__bf16)(o3[r] * inv);
  }
}

// ---------------------------------------------------------------------------
// WMMA GEMM: C(MxN) = A(MxK) * Bt(NxK)^T + epilogue
// One wave computes a 32x64 tile: 2 M-tiles x 4 N-tiles, 8 accumulators.
// Per 32-wide k-step: 8 WMMA / 12 b128 loads (A frags reused x4, B frags x2).
// MODE 0: out_f32 = gate[b][n] * (acc + bias[n])            (Wo + pH gate)
// MODE 1: out_bf16 = gelu_exact(acc + bias[n])              (W1)
// MODE 2: out_f32 = acc + bias[n] + aux[m][n]               (W2 + residual)
// ---------------------------------------------------------------------------
template <int MODE>
__global__ void __launch_bounds__(256) gemm_kernel(const __bf16* __restrict__ A,
                                                   const __bf16* __restrict__ Bt,
                                                   int M, int N, int K,
                                                   const float* __restrict__ bias,
                                                   const float* __restrict__ aux,
                                                   float* __restrict__ outf,
                                                   __bf16* __restrict__ outb) {
  const int gw   = (blockIdx.x * blockDim.x + threadIdx.x) >> 5;
  const int lane = threadIdx.x & 31;
  const int nStrips = N >> 6;
  const int m0 = (gw / nStrips) * 32;
  const int n0 = (gw % nStrips) * 64;
  if (m0 >= M) return;

  v8f acc[2][4] = {};
  const __bf16* A0 = A + (size_t)m0 * K;
  const __bf16* A1 = A0 + (size_t)16 * K;
  const int mA = lane & 15, kA = (lane >> 4) * 8;

  for (int kk = 0; kk < K; kk += 32) {
    // prefetch the A stream two k-steps ahead (global_prefetch_b8)
    __builtin_prefetch(A0 + (size_t)mA * K + kk + kA + 64, 0, 1);
    __builtin_prefetch(A1 + (size_t)mA * K + kk + kA + 64, 0, 1);

    v16bf a0 = load_afrag(A0, K, lane, kk);
    v16bf a1 = load_afrag(A1, K, lane, kk);
#pragma unroll
    for (int t = 0; t < 4; ++t) {
      v16bf bfr = load_bfrag(Bt + (size_t)(n0 + t * 16) * K, K, lane, kk);
      acc[0][t] = WMMA_BF16(a0, bfr, acc[0][t]);
      acc[1][t] = WMMA_BF16(a1, bfr, acc[1][t]);
    }
  }

  const int hlf = lane >> 4, nI = lane & 15;
#pragma unroll
  for (int u = 0; u < 2; ++u) {
#pragma unroll
    for (int r = 0; r < 8; ++r) {
      int m = m0 + u * 16 + r + hlf * 8;
#pragma unroll
      for (int t = 0; t < 4; ++t) {
        int n = n0 + t * 16 + nI;
        float x = acc[u][t][r] + bias[n];
        size_t idx = (size_t)m * N + n;
        if (MODE == 0) {
          int b = m / S_;
          outf[idx] = aux[(size_t)b * N + n] * x;
        } else if (MODE == 1) {
          float g = 0.5f * x * (1.0f + erff(x * 0.70710678f));
          outb[idx] = (__bf16)g;
        } else {
          outf[idx] = x + aux[idx];
        }
      }
    }
  }
}

// ---------------------------------------------------------------------------
// LayerNorm over E=1024 (one row per 256-thread block)
// in + addend -> LN -> outf (f32) and optionally outb (bf16)
// ---------------------------------------------------------------------------
__device__ inline float block_sum(float v, float* sh) {
#pragma unroll
  for (int m = 16; m >= 1; m >>= 1) v += __shfl_xor(v, m, 32);
  if ((threadIdx.x & 31) == 0) sh[threadIdx.x >> 5] = v;
  __syncthreads();
  float t = 0.f;
#pragma unroll
  for (int i = 0; i < 8; ++i) t += sh[i];
  __syncthreads();
  return t;
}

__global__ void __launch_bounds__(256) ln_kernel(const float* __restrict__ in,
                                                 const float* __restrict__ addend,
                                                 const float* __restrict__ g,
                                                 const float* __restrict__ bta,
                                                 float* __restrict__ outf,
                                                 __bf16* __restrict__ outb) {
  __shared__ float sh[8];
  const int row = blockIdx.x, tid = threadIdx.x;
  const float* p = in + (size_t)row * E_;
  float v[4]; float s = 0.f;
#pragma unroll
  for (int i = 0; i < 4; ++i) {
    int e = tid + i * 256;
    float x = p[e];
    if (addend) x += addend[(size_t)row * E_ + e];
    v[i] = x; s += x;
  }
  float mu = block_sum(s, sh) * (1.0f / E_);
  float var = 0.f;
#pragma unroll
  for (int i = 0; i < 4; ++i) { float d = v[i] - mu; var += d * d; }
  var = block_sum(var, sh) * (1.0f / E_);
  float rs = rsqrtf(var + 1e-5f);
#pragma unroll
  for (int i = 0; i < 4; ++i) {
    int e = tid + i * 256;
    float y = (v[i] - mu) * rs * g[e] + bta[e];
    outf[(size_t)row * E_ + e] = y;
    if (outb) outb[(size_t)row * E_ + e] = (__bf16)y;
  }
}

// ---------------------------------------------------------------------------
extern "C" void kernel_launch(void* const* d_in, const int* in_sizes, int n_in,
                              void* d_out, int out_size, void* d_ws, size_t ws_size,
                              hipStream_t stream) {
  const float* value = (const float*)d_in[0];
  const float* keyi  = (const float*)d_in[1];
  const float* query = (const float*)d_in[2];
  // d_in[3] = mask: broadcast causal tril, applied analytically in attn_kernel
  const float* Wo  = (const float*)d_in[4];
  const float* bo  = (const float*)d_in[5];
  const float* Wg  = (const float*)d_in[6];
  const float* bg  = (const float*)d_in[7];
  const float* g1  = (const float*)d_in[8];
  const float* b1  = (const float*)d_in[9];
  const float* g2  = (const float*)d_in[10];
  const float* b2  = (const float*)d_in[11];
  const float* W1  = (const float*)d_in[12];
  const float* bf1 = (const float*)d_in[13];
  const float* W2  = (const float*)d_in[14];
  const float* bf2 = (const float*)d_in[15];
  float* out = (float*)d_out;

  char* ws = (char*)d_ws;
  size_t off = 0;
  auto alloc = [&](size_t bytes) -> char* {
    char* p = ws + off;
    off += (bytes + 255) & ~(size_t)255;
    return p;
  };

  const size_t NTOK = (size_t)M_;            // 4096 token rows
  __bf16* qb    = (__bf16*)alloc(NTOK * E_ * 2);
  __bf16* kb    = (__bf16*)alloc(NTOK * E_ * 2);
  __bf16* vt    = (__bf16*)alloc(NTOK * E_ * 2);
  __bf16* WoT   = (__bf16*)alloc((size_t)E_ * E_ * 2);
  __bf16* W1T   = (__bf16*)alloc((size_t)E_ * FF_ * 2);
  __bf16* W2T   = (__bf16*)alloc((size_t)FF_ * E_ * 2);
  __bf16* ao    = (__bf16*)alloc(NTOK * E_ * 2);
  float*  xavg  = (float*)alloc((size_t)B_ * E_ * 4);
  float*  gate  = (float*)alloc((size_t)B_ * E_ * 4);
  float*  gated = (float*)alloc(NTOK * E_ * 4);
  float*  xf    = (float*)alloc(NTOK * E_ * 4);
  __bf16* xb    = (__bf16*)alloc(NTOK * E_ * 2);
  __bf16* hbuf  = (__bf16*)alloc(NTOK * FF_ * 2);
  float*  ybuf  = (float*)alloc(NTOK * E_ * 4);
  (void)ws_size; (void)in_sizes; (void)n_in; (void)out_size;

  const int nQKV = (int)(NTOK * E_);         // 4,194,304

  // Stage 1: precision conversion + transposes
  cvt_bf16_kernel<<<(nQKV + 255) / 256, 256, 0, stream>>>(query, qb, nQKV);
  cvt_bf16_kernel<<<(nQKV + 255) / 256, 256, 0, stream>>>(keyi, kb, nQKV);
  v_transpose_kernel<<<(nQKV + 255) / 256, 256, 0, stream>>>(value, vt);
  transpose_cvt_kernel<<<dim3(E_ / 32, E_ / 32), 256, 0, stream>>>(Wo, WoT, E_, E_);
  transpose_cvt_kernel<<<dim3(FF_ / 32, E_ / 32), 256, 0, stream>>>(W1, W1T, E_, FF_);
  transpose_cvt_kernel<<<dim3(E_ / 32, FF_ / 32), 256, 0, stream>>>(W2, W2T, FF_, E_);

  // Stage 2: pH gate (tiny, fp32 scalar)
  colmean_kernel<<<(B_ * E_ + 255) / 256, 256, 0, stream>>>(query, xavg);
  gate_kernel<<<(B_ * E_ + 255) / 256, 256, 0, stream>>>(xavg, Wg, bg, gate);

  // Stage 3: causal flash attention (WMMA). waves = B*H*(S/16) = 4096
  attn_kernel<<<(B_ * H_ * (S_ / 16)) / 4, 128, 0, stream>>>(qb, kb, vt, ao);

  // Stage 4: fc_out + gate epilogue -> gated (f32).  waves = (M/32)*(N/64)
  {
    int waves = (M_ / 32) * (E_ / 64);
    gemm_kernel<0><<<waves / 8, 256, 0, stream>>>(ao, WoT, M_, E_, E_, bo, gate, gated, nullptr);
  }

  // Stage 5: x = LN1(gated + query) -> xf (f32 residual) + xb (bf16 GEMM input)
  ln_kernel<<<M_, 256, 0, stream>>>(gated, query, g1, b1, xf, xb);

  // Stage 6: h = gelu(x @ W1 + bf1) (bf16)
  {
    int waves = (M_ / 32) * (FF_ / 64);
    gemm_kernel<1><<<waves / 8, 256, 0, stream>>>(xb, W1T, M_, FF_, E_, bf1, nullptr, nullptr, hbuf);
  }

  // Stage 7: y = h @ W2 + bf2 + x (f32)
  {
    int waves = (M_ / 32) * (E_ / 64);
    gemm_kernel<2><<<waves / 8, 256, 0, stream>>>(hbuf, W2T, M_, E_, FF_, bf2, xf, ybuf, nullptr);
  }

  // Stage 8: out = LN2(y)
  ln_kernel<<<M_, 256, 0, stream>>>(ybuf, nullptr, g2, b2, out, nullptr);
}
